// FeatureFusionBlock_v2_75479755260682
// MI455X (gfx1250) — compile-verified, matrix operand order checked by reading, not run on hardware
//
#include <hip/hip_runtime.h>
#include <hip/hip_bf16.h>
#include <math.h>

// ---------------------------------------------------------------------------
// FeatureFusionBlock_v2 on gfx1250 (CDNA5, wave32, WMMA bf16 16x16x32)
// B=4, C_in=128 (concat -> 256), H=W=64, F=256
// ---------------------------------------------------------------------------

typedef __attribute__((ext_vector_type(16))) __bf16 bf16x16;
typedef __attribute__((ext_vector_type(8)))  float  f32x8;

#define HW    4096
#define IMG_H 64
#define IMG_W 64
#define CF    256
#define NB    4

__device__ __forceinline__ unsigned short f2bf(float f) {
  unsigned int u = __float_as_uint(f);
  u += 0x7FFFu + ((u >> 16) & 1u);          // round-to-nearest-even
  return (unsigned short)(u >> 16);
}

__device__ __forceinline__ __bf16 bfbits(unsigned short u) {
  union { unsigned short s; __bf16 b; } cv; cv.s = u; return cv.b;
}

// K index inside a 16-bit A/B fragment per ISA layout (16x32, wave32):
// lanes 0-15: VGPR0..3 -> K 0..7, VGPR4..7 -> K 16..23 ; lanes 16-31: +8
__device__ __forceinline__ int frag_k(int e, int lane) {
  int j = e >> 1;
  int k = (j < 4 ? (j << 1) : 16 + ((j - 4) << 1));
  if (lane & 16) k += 8;
  return k + (e & 1);
}

// Channel-concat aware activation load: c<128 from x0, else x1.
// bstr = per-batch stride in floats (128*HW for raw concat inputs,
// 256*HW with x1 = buf + 128*HW for internal [b][256][hw] buffers).
__device__ __forceinline__ float loadX(const float* __restrict__ x0,
                                       const float* __restrict__ x1,
                                       size_t bstr, int b, int c, int m,
                                       int relu) {
  const float* p = (c < 128) ? x0 : x1;
  int cc = (c < 128) ? c : (c - 128);
  float v = p[(size_t)b * bstr + (size_t)cc * HW + m];
  return relu ? fmaxf(v, 0.f) : v;
}

// ---------------------------------------------------------------------------
// Fused Q/K/V 1x1-conv GEMM:  Y[o,m] = W[o,c] * relu(X[c,m]) + bias[o]
// Workgroup tile: 64 out-ch x 128 positions, 8 waves (2x4), per wave 32x32.
// ---------------------------------------------------------------------------
__global__ void __launch_bounds__(256)
ffb_qkv_gemm(const float* __restrict__ x0, const float* __restrict__ x1,
             unsigned long long bstr, int relu_in,
             const float* __restrict__ Wq, const float* __restrict__ Bq,
             const float* __restrict__ Wk, const float* __restrict__ Bk,
             const float* __restrict__ Wv, const float* __restrict__ Bv,
             float* __restrict__ Qo, float* __restrict__ Ko,
             float* __restrict__ Vo) {
  __shared__ unsigned short sW[3][64][33];
  __shared__ unsigned short sX[128][33];

  const int tid = threadIdx.x, lane = tid & 31, wid = tid >> 5;
  const int ci = wid >> 2, pi = wid & 3;               // wave grid 2 x 4
  const int m0 = blockIdx.x * 128;                     // position tile
  const int c0 = blockIdx.y * 64;                      // out-channel tile
  const int b  = blockIdx.z;

  const f32x8 zero = {0.f,0.f,0.f,0.f,0.f,0.f,0.f,0.f};
  f32x8 acc[3][2][2];
  #pragma unroll
  for (int mat = 0; mat < 3; ++mat)
    #pragma unroll
    for (int mb = 0; mb < 2; ++mb)
      #pragma unroll
      for (int pb = 0; pb < 2; ++pb) acc[mat][mb][pb] = zero;

  for (int k0 = 0; k0 < CF; k0 += 32) {
    // Stage weights (3 matrices x 64x32) as bf16
    for (int idx = tid; idx < 3 * 64 * 32; idx += 256) {
      int mat = idx >> 11, rem = idx & 2047, r = rem >> 5, cc = rem & 31;
      const float* Wm = (mat == 0) ? Wq : (mat == 1) ? Wk : Wv;
      sW[mat][r][cc] = f2bf(Wm[(size_t)(c0 + r) * CF + k0 + cc]);
    }
    // Stage activations transposed [pos][k] (global reads coalesced over m)
    for (int idx = tid; idx < 128 * 32; idx += 256) {
      int p = idx & 127, cc = idx >> 7;
      sX[p][cc] = f2bf(loadX(x0, x1, (size_t)bstr, b, k0 + cc, m0 + p, relu_in));
    }
    // Prefetch next K-slice of X into caches (global_prefetch_b8)
    if (k0 + 32 < CF) {
      int cn = k0 + 32 + (tid >> 3);
      const float* p = (cn < 128) ? x0 : x1;
      int cc = (cn < 128) ? cn : cn - 128;
      __builtin_prefetch(&p[(size_t)b * bstr + (size_t)cc * HW + m0 +
                            ((tid & 7) << 4)], 0, 1);
    }
    __syncthreads();

    bf16x16 fb[2];
    #pragma unroll
    for (int pb = 0; pb < 2; ++pb) {
      const int n = pi * 32 + pb * 16 + (lane & 15);
      #pragma unroll
      for (int e = 0; e < 16; ++e) fb[pb][e] = bfbits(sX[n][frag_k(e, lane)]);
    }
    #pragma unroll
    for (int mat = 0; mat < 3; ++mat) {
      #pragma unroll
      for (int mb = 0; mb < 2; ++mb) {
        bf16x16 fa;
        const int rr = ci * 32 + mb * 16 + (lane & 15);
        #pragma unroll
        for (int e = 0; e < 16; ++e) fa[e] = bfbits(sW[mat][rr][frag_k(e, lane)]);
        #pragma unroll
        for (int pb = 0; pb < 2; ++pb)
          acc[mat][mb][pb] = __builtin_amdgcn_wmma_f32_16x16x32_bf16(
              false, fa, false, fb[pb], (short)0, acc[mat][mb][pb],
              false, false);
      }
    }
    __syncthreads();
  }

  // Epilogue: + bias, store fp32 (C layout: VGPR r -> M=r / M=r+8 by lane half)
  const float* biases[3] = {Bq, Bk, Bv};
  float* outs[3] = {Qo, Ko, Vo};
  #pragma unroll
  for (int mat = 0; mat < 3; ++mat)
    #pragma unroll
    for (int mb = 0; mb < 2; ++mb)
      #pragma unroll
      for (int pb = 0; pb < 2; ++pb) {
        const int col = m0 + pi * 32 + pb * 16 + (lane & 15);
        #pragma unroll
        for (int r = 0; r < 8; ++r) {
          const int ch = c0 + ci * 32 + mb * 16 + r + ((lane & 16) ? 8 : 0);
          outs[mat][((size_t)b * CF + ch) * HW + col] =
              acc[mat][mb][pb][r] + biases[mat][ch];
        }
      }
}

// ---------------------------------------------------------------------------
// Windowed attention: softmax(q * (k3x3 + rel)) . v3x3, fused residual adds.
// One (b, c, 16x16 spatial tile) per workgroup; K/V halo = conv bias.
// ---------------------------------------------------------------------------
__global__ void __launch_bounds__(256)
ffb_attn(const float* __restrict__ Qv, const float* __restrict__ Kv,
         const float* __restrict__ Vv,
         const float* __restrict__ Bk, const float* __restrict__ Bv,
         const float* __restrict__ rh, const float* __restrict__ rw,
         const float* __restrict__ r0a, const float* __restrict__ r0b,
         const float* __restrict__ r1, float* __restrict__ O) {
  __shared__ float sK[18][18];
  __shared__ float sV[18][18];
  const int tid = threadIdx.x;
  const int c = blockIdx.y, b = blockIdx.z;
  const int th = (blockIdx.x >> 2) << 4, tw = (blockIdx.x & 3) << 4;
  const size_t base = ((size_t)b * CF + c) * HW;
  const float kb = Bk[c], vb = Bv[c];

  for (int idx = tid; idx < 18 * 18; idx += 256) {
    const int r = idx / 18, q = idx % 18;
    const int hh = th + r - 1, ww = tw + q - 1;
    const bool in = (hh >= 0) & (hh < IMG_H) & (ww >= 0) & (ww < IMG_W);
    const size_t g = base + (size_t)(in ? hh : 0) * IMG_W + (in ? ww : 0);
    sK[r][q] = in ? Kv[g] : kb;   // zero-padded conv1x1 => bias at border
    sV[r][q] = in ? Vv[g] : vb;
  }
  __syncthreads();

  const int ly = tid >> 4, lx = tid & 15;
  const int h = th + ly, w = tw + lx;
  const float q = Qv[base + (size_t)h * IMG_W + w];

  float e[9], mx = -3.4e38f;
  #pragma unroll
  for (int ky = 0; ky < 3; ++ky)
    #pragma unroll
    for (int kx = 0; kx < 3; ++kx) {
      const float kk = sK[ly + ky][lx + kx] + ((c < 128) ? rh[ky] : rw[kx]);
      const float t = q * kk;
      e[ky * 3 + kx] = t;
      mx = fmaxf(mx, t);
    }
  float s = 0.f, o = 0.f;
  #pragma unroll
  for (int ky = 0; ky < 3; ++ky)
    #pragma unroll
    for (int kx = 0; kx < 3; ++kx) {
      const float wgt = __expf(e[ky * 3 + kx] - mx);
      s += wgt;
      o += wgt * sV[ly + ky][lx + kx];
    }
  o /= s;

  const int m = h * IMG_W + w;
  if (r0a) o += loadX(r0a, r0b, (size_t)128 * HW, b, c, m, 0);  // concat resid
  if (r1)  o += r1[base + m];                                   // buffer resid
  O[base + m] = o;
}

// ---------------------------------------------------------------------------
// 3x3 conv head as 9-tap WMMA GEMM, fused tanh + pixel_shuffle(2) writeout.
// ---------------------------------------------------------------------------
__global__ void __launch_bounds__(256)
ffb_conv3x3_ps(const float* __restrict__ X, const float* __restrict__ Wc,
               const float* __restrict__ Bc, int OC, float* __restrict__ out) {
  __shared__ unsigned short sW[64][33];
  __shared__ unsigned short sX[128][33];
  const int tid = threadIdx.x, lane = tid & 31, wid = tid >> 5;
  const int ci = wid >> 2, pi = wid & 3;
  const int m0 = blockIdx.x * 128, c0 = blockIdx.y * 64, b = blockIdx.z;

  const f32x8 zero = {0.f,0.f,0.f,0.f,0.f,0.f,0.f,0.f};
  f32x8 acc[2][2];
  #pragma unroll
  for (int mb = 0; mb < 2; ++mb)
    #pragma unroll
    for (int pb = 0; pb < 2; ++pb) acc[mb][pb] = zero;

  for (int t = 0; t < 9; ++t) {
    const int ky = t / 3 - 1, kx = t % 3 - 1;
    for (int k0 = 0; k0 < CF; k0 += 32) {
      for (int idx = tid; idx < 64 * 32; idx += 256) {
        const int r = idx >> 5, cc = idx & 31, oc = c0 + r;
        const float wv =
            (oc < OC) ? Wc[((size_t)oc * CF + (k0 + cc)) * 9 + t] : 0.f;
        sW[r][cc] = f2bf(wv);
      }
      for (int idx = tid; idx < 128 * 32; idx += 256) {
        const int p = idx & 127, cc = idx >> 7;
        const int m = m0 + p, h = m >> 6, w = m & 63;
        const int hh = h + ky, ww = w + kx;
        const bool in = (hh >= 0) & (hh < IMG_H) & (ww >= 0) & (ww < IMG_W);
        const float xv =
            in ? X[((size_t)b * CF + (k0 + cc)) * HW + hh * IMG_W + ww] : 0.f;
        sX[p][cc] = f2bf(xv);
      }
      __syncthreads();

      bf16x16 fb[2];
      #pragma unroll
      for (int pb = 0; pb < 2; ++pb) {
        const int n = pi * 32 + pb * 16 + (lane & 15);
        #pragma unroll
        for (int e = 0; e < 16; ++e) fb[pb][e] = bfbits(sX[n][frag_k(e, lane)]);
      }
      #pragma unroll
      for (int mb = 0; mb < 2; ++mb) {
        bf16x16 fa;
        const int rr = ci * 32 + mb * 16 + (lane & 15);
        #pragma unroll
        for (int e = 0; e < 16; ++e) fa[e] = bfbits(sW[rr][frag_k(e, lane)]);
        #pragma unroll
        for (int pb = 0; pb < 2; ++pb)
          acc[mb][pb] = __builtin_amdgcn_wmma_f32_16x16x32_bf16(
              false, fa, false, fb[pb], (short)0, acc[mb][pb], false, false);
      }
      __syncthreads();
    }
  }

  // tanh + pixel-shuffle(2): oc = co*4 + r*2 + s
  const int nco = OC >> 2;
  #pragma unroll
  for (int mb = 0; mb < 2; ++mb)
    #pragma unroll
    for (int pb = 0; pb < 2; ++pb) {
      const int col = m0 + pi * 32 + pb * 16 + (lane & 15);
      const int h = col >> 6, w = col & 63;
      #pragma unroll
      for (int r = 0; r < 8; ++r) {
        const int oc = c0 + ci * 32 + mb * 16 + r + ((lane & 16) ? 8 : 0);
        if (oc < OC) {
          const float val = tanhf(acc[mb][pb][r] + Bc[oc]);
          const int co = oc >> 2, rr = (oc >> 1) & 1, ss = oc & 1;
          out[(((size_t)b * nco + co) * (2 * IMG_H) + (2 * h + rr)) *
                  (2 * IMG_W) + (2 * w + ss)] = val;
        }
      }
    }
}

// ---------------------------------------------------------------------------
// Orchestration
// ---------------------------------------------------------------------------
extern "C" void kernel_launch(void* const* d_in, const int* in_sizes, int n_in,
                              void* d_out, int out_size, void* d_ws,
                              size_t ws_size, hipStream_t stream) {
  (void)in_sizes; (void)n_in; (void)out_size; (void)ws_size;
  const float* dt    = (const float*)d_in[0];
  const float* upt   = (const float*)d_in[1];
  const float* dt1   = (const float*)d_in[2];
  const float* dt2   = (const float*)d_in[3];
  const float* q_w   = (const float*)d_in[4];
  const float* q_b   = (const float*)d_in[5];
  const float* k_w   = (const float*)d_in[6];
  const float* k_b   = (const float*)d_in[7];
  const float* v_w   = (const float*)d_in[8];
  const float* v_b   = (const float*)d_in[9];
  const float* rel_h = (const float*)d_in[10];
  const float* rel_w = (const float*)d_in[11];
  const float* up1w  = (const float*)d_in[12];
  const float* up1b  = (const float*)d_in[13];
  const float* up2w  = (const float*)d_in[14];
  const float* up2b  = (const float*)d_in[15];
  float* out = (float*)d_out;

  const size_t Nact = (size_t)NB * CF * HW;  // 4*256*4096 floats = 16 MB
  float* ws = (float*)d_ws;
  float* Qb = ws + 0 * Nact;
  float* Kb = ws + 1 * Nact;
  float* Vb = ws + 2 * Nact;
  float* T0 = ws + 3 * Nact;
  float* T1 = ws + 4 * Nact;
  float* T2 = ws + 5 * Nact;

  const dim3 blk(256);
  const dim3 gG(32, 4, NB);       // GEMM: 32 pos tiles x 4 ch tiles x B
  const dim3 gA(16, CF, NB);      // attn: 16 spatial tiles x 256 ch x B
  const unsigned long long bsC = (unsigned long long)128 * HW;  // concat inputs
  const unsigned long long bsB = (unsigned long long)256 * HW;  // ws buffers

  auto qkv = [&](const float* a, const float* a2, unsigned long long bs, int i) {
    ffb_qkv_gemm<<<gG, blk, 0, stream>>>(
        a, a2, bs, 1,
        q_w + (size_t)i * CF * CF, q_b + (size_t)i * CF,
        k_w + (size_t)i * CF * CF, k_b + (size_t)i * CF,
        v_w + (size_t)i * CF * CF, v_b + (size_t)i * CF, Qb, Kb, Vb);
  };
  auto attn = [&](int i, const float* r0A, const float* r0B, const float* R1,
                  float* O) {
    ffb_attn<<<gA, blk, 0, stream>>>(
        Qb, Kb, Vb, k_b + (size_t)i * CF, v_b + (size_t)i * CF,
        rel_h + (size_t)i * 3, rel_w + (size_t)i * 3, r0A, r0B, R1, O);
  };

  // RAU 0 on concat(dt, upt): T1 = attn1(relu(attn0(relu(x)))) + x
  qkv(dt, upt, bsC, 0);
  attn(0, nullptr, nullptr, nullptr, T0);
  qkv(T0, T0 + 128 * HW, bsB, 1);
  attn(1, dt, upt, nullptr, T1);

  // RAU 1 on concat(dt_1, dt_2), fused sum with RAU0: T2 = rau1 + T1
  qkv(dt1, dt2, bsC, 2);
  attn(2, nullptr, nullptr, nullptr, T0);
  qkv(T0, T0 + 128 * HW, bsB, 3);
  attn(3, dt1, dt2, T1, T2);

  // RAU 2 on T2: T1 = attn5(relu(attn4(relu(T2)))) + T2
  qkv(T2, T2 + 128 * HW, bsB, 4);
  attn(4, nullptr, nullptr, nullptr, T0);
  qkv(T0, T0 + 128 * HW, bsB, 5);
  attn(5, nullptr, nullptr, T2, T1);

  // Heads: depth (4 ch -> [4,1,128,128]) then feature (256 ch -> [4,64,128,128])
  ffb_conv3x3_ps<<<dim3(32, 1, NB), blk, 0, stream>>>(T1, up1w, up1b, 4, out);
  ffb_conv3x3_ps<<<dim3(32, 4, NB), blk, 0, stream>>>(T1, up2w, up2b, CF,
                                                      out + (size_t)NB * 1 *
                                                      (2 * IMG_H) * (2 * IMG_W));
}
